// SequentialOctave_87660282511418
// MI455X (gfx1250) — matrix operand order repaired and checked by reading, not compiled, not run on hardware
//
#include <hip/hip_runtime.h>
#include <hip/hip_bf16.h>

// ---------------- problem constants ----------------
#define B_   512
#define T_   256
#define D_   64
#define H_   128
#define Z_   64
#define C_   32
#define G_   512            // 4*H
#define KTOT 192            // D + H (fused [x|h] K dimension)
#define MT   16             // batch rows per workgroup
#define NWG  (B_/MT)        // 32 workgroups
#define KS_GATE 6           // 192/32 K-steps for gate GEMM
#define KS_OUT  4           // 128/32 K-steps for fc2 GEMM

typedef __attribute__((ext_vector_type(16))) _Float16 v16h;
typedef __attribute__((ext_vector_type(8)))  float    v8f;

// packed-weight scratch layout (units: halfs)
#define ENC_FRAG0 0
#define DEC_FRAG0 (192*512)
#define FC2_FRAG0 (384*512)
#define WS_HALFS  (400*512)   // ~400 KB of d_ws

// ============================================================
// Pre-kernel: pack f32 weights into f16 WMMA-B fragment order.
// B operand (32x16 KxN f16): lane l -> N = l&15, K = (l>=16)*16 + h,
// h = 0..15 stored contiguously (32B per lane, 1024B per fragment).
// ============================================================
__global__ __launch_bounds__(256) void pack_weights(
    const float* __restrict__ gWih, const float* __restrict__ gWhh,
    const float* __restrict__ rWih, const float* __restrict__ rWhh,
    const float* __restrict__ rfc2W, _Float16* __restrict__ ws) {
  int gid = blockIdx.x * 256 + threadIdx.x;
  if (gid >= WS_HALFS) return;
  int frag_g = gid >> 9;          // global fragment id (512 halfs each)
  int e      = gid & 511;         // element = lane*16 + h
  int lane = e >> 4, h = e & 15;
  int Khi = (lane >> 4) << 4;     // 0 or 16
  float val;
  if (frag_g < 384) {             // gate weights: [W_ih ; W_hh] K=192, N=512
    int set = frag_g / 192;       // 0 = encoder, 1 = decoder
    int f   = frag_g % 192;
    int ntile = f / KS_GATE, kstep = f % KS_GATE;
    int K = kstep * 32 + Khi + h;
    int N = ntile * 16 + (lane & 15);
    const float* Wih = set ? rWih : gWih;   // [512, 64]  (row n, col k)
    const float* Whh = set ? rWhh : gWhh;   // [512, 128]
    val = (K < D_) ? Wih[N * D_ + K] : Whh[N * H_ + (K - D_)];
  } else {                        // r_fc2_W: [64, 128], K=128, N=64
    int f = frag_g - 384;
    int ntile = f / KS_OUT, kstep = f % KS_OUT;
    int K = kstep * 32 + Khi + h;
    int N = ntile * 16 + (lane & 15);
    val = rfc2W[N * H_ + K];
  }
  ws[(size_t)frag_g * 512 + (size_t)lane * 16 + h] = (_Float16)val;
}

// ============================================================
// Fragment loaders (wave32; contiguous k order shared by A and B, not the ISA table order)
// ============================================================
// A (16x32 f16): lane l -> M = l&15; VGPR v:
//   K = kbase + (l>=16)*16 + v*2 + pair
__device__ inline v16h load_a_frag(const unsigned int* a_u, int lane, int kbase) {
  union { v16h v; unsigned int u[8]; } a;
  int m = lane & 15, hi = lane >> 4;
  int row = m * (KTOT / 2);                 // uints per row
#pragma unroll
  for (int p = 0; p < 8; ++p) {
    int k = kbase + hi * 16 + p * 2;  // even
    a.u[p] = a_u[row + (k >> 1)];
  }
  return a.v;
}

// B (32x16 f16): each lane's 16 halfs are contiguous in packed scratch
__device__ inline v16h load_b_frag(const unsigned int* wb, int frag, int lane) {
  union { v16h v; unsigned int u[8]; } b;
  const unsigned int* p = wb + (size_t)frag * 256 + lane * 8;   // 32B/lane
#pragma unroll
  for (int i = 0; i < 8; ++i) b.u[i] = p[i];
  return b.v;
}

__device__ inline float sigf(float x) { return 1.0f / (1.0f + __expf(-x)); }

// gates[MT x 512] = A(16x192) @ Wpacked(192x512); wave w owns N tiles w*4..w*4+3
__device__ inline void gate_gemm(const unsigned int* a_u, const unsigned int* wb,
                                 float* gates, int w, int lane) {
  v8f acc[4] = {};
#pragma unroll
  for (int ks = 0; ks < KS_GATE; ++ks) {
    v16h a = load_a_frag(a_u, lane, ks * 32);
#pragma unroll
    for (int nt = 0; nt < 4; ++nt) {
      v16h b = load_b_frag(wb, (w * 4 + nt) * KS_GATE + ks, lane);
      acc[nt] = __builtin_amdgcn_wmma_f32_16x16x32_f16(
          false, a, false, b, (short)0, acc[nt], false, false);
    }
  }
  // C/D layout: VGPR r, lane l -> M = r + (l>=16)*8, N = l&15
  int mhi = (lane >> 4) * 8, nl = lane & 15;
#pragma unroll
  for (int nt = 0; nt < 4; ++nt) {
    int n = (w * 4 + nt) * 16 + nl;
#pragma unroll
    for (int r = 0; r < 8; ++r) gates[(mhi + r) * G_ + n] = acc[nt][r];
  }
}

// ============================================================
// Main persistent kernel: one WG per 16-row batch tile, runs
// encoder scan -> VAE head -> autoregressive decoder scan.
// ============================================================
__global__ __launch_bounds__(256) void vrnn_main(
    const float* __restrict__ x,   const float* __restrict__ I,
    const float* __restrict__ eps,
    const float* __restrict__ g_b_ih, const float* __restrict__ g_b_hh,
    const float* __restrict__ g_fc_W, const float* __restrict__ g_fc_b,
    const float* __restrict__ r_fc1_W, const float* __restrict__ r_fc1_b,
    const float* __restrict__ r_b_ih, const float* __restrict__ r_b_hh,
    const float* __restrict__ r_fc2_b,
    const unsigned int* __restrict__ wfrag,   // packed f16 fragments
    float* __restrict__ out) {
  __shared__ float gates[MT * G_];       // 32 KB
  __shared__ float hbuf[MT * H_];        // 8 KB
  __shared__ float cbuf[MT * H_];        // 8 KB
  __shared__ float bias[G_];             // 2 KB
  __shared__ unsigned int a_u[MT * (KTOT / 2)];  // 6 KB  ([x|h] in f16)
  _Float16* a_h = (_Float16*)a_u;

  const int tid = threadIdx.x, w = tid >> 5, lane = tid & 31;
  const int m0 = blockIdx.x * MT;
  const size_t MU_OFF = (size_t)B_ * T_ * D_;
  const size_t LV_OFF = MU_OFF + (size_t)B_ * Z_;

  const unsigned int* encW = wfrag + ENC_FRAG0 / 2;
  const unsigned int* decW = wfrag + DEC_FRAG0 / 2;
  const unsigned int* fc2W = wfrag + FC2_FRAG0 / 2;

  // ---- encoder init: h = c = 0, combined bias ----
  if (tid < G_) bias[tid] = g_b_ih[tid] + g_b_hh[tid];
  for (int idx = tid; idx < MT * H_; idx += 256) {
    hbuf[idx] = 0.f; cbuf[idx] = 0.f;
    int m = idx >> 7, j = idx & 127;
    a_h[m * KTOT + D_ + j] = (_Float16)0.f;
  }
  __syncthreads();

  // ---------------- encoder scan ----------------
  for (int t = 0; t < T_; ++t) {
    for (int idx = tid; idx < MT * D_; idx += 256) {  // stage x_t -> f16
      int m = idx >> 6, d = idx & 63;
      size_t src = ((size_t)(m0 + m) * T_ + t) * D_ + d;
      a_h[m * KTOT + d] = (_Float16)x[src];
      if (t + 1 < T_) __builtin_prefetch(&x[src + D_], 0, 0);
    }
    __syncthreads();
    gate_gemm(a_u, encW, gates, w, lane);
    __syncthreads();
    for (int idx = tid; idx < MT * H_; idx += 256) {  // LSTM cell (f32)
      int m = idx >> 7, j = idx & 127;
      const float* gm = &gates[m * G_];
      float ig = gm[j]          + bias[j];
      float fg = gm[H_ + j]     + bias[H_ + j];
      float gg = gm[2 * H_ + j] + bias[2 * H_ + j];
      float og = gm[3 * H_ + j] + bias[3 * H_ + j];
      float c = sigf(fg) * cbuf[idx] + sigf(ig) * tanhf(gg);
      float h = sigf(og) * tanhf(c);
      cbuf[idx] = c; hbuf[idx] = h;
      a_h[m * KTOT + D_ + j] = (_Float16)h;
    }
    __syncthreads();
  }

  // ---------------- VAE head ----------------
  // [h_enc | I] @ g_fc_W^T + b  -> gates[m][0..127]
  for (int idx = tid; idx < MT * 2 * Z_; idx += 256) {
    int m = idx >> 7, n = idx & 127;
    float acc = g_fc_b[n];
    const float* wr = &g_fc_W[n * (H_ + C_)];
    for (int k = 0; k < H_; ++k) acc += hbuf[m * H_ + k] * wr[k];
    const float* Ir = &I[(size_t)(m0 + m) * C_];
    for (int k = 0; k < C_; ++k) acc += Ir[k] * wr[H_ + k];
    gates[m * G_ + n] = acc;
  }
  __syncthreads();
  for (int idx = tid; idx < MT * Z_; idx += 256) {   // reparameterize
    int m = idx >> 6, zn = idx & 63;
    float mu = gates[m * G_ + zn];
    float lv = gates[m * G_ + Z_ + zn];
    float z  = mu + eps[(size_t)(m0 + m) * Z_ + zn] * __expf(0.5f * lv);
    gates[m * G_ + 128 + zn] = z;
    out[MU_OFF + (size_t)(m0 + m) * Z_ + zn] = mu;
    out[LV_OFF + (size_t)(m0 + m) * Z_ + zn] = lv;
  }
  __syncthreads();
  // hd = [z | I] @ r_fc1_W^T + b ; reset decoder state
  for (int idx = tid; idx < MT * H_; idx += 256) {
    int m = idx >> 7, n = idx & 127;
    float acc = r_fc1_b[n];
    const float* wr = &r_fc1_W[n * (Z_ + C_)];
    for (int k = 0; k < Z_; ++k) acc += gates[m * G_ + 128 + k] * wr[k];
    const float* Ir = &I[(size_t)(m0 + m) * C_];
    for (int k = 0; k < C_; ++k) acc += Ir[k] * wr[Z_ + k];
    hbuf[idx] = acc; cbuf[idx] = 0.f;
    a_h[m * KTOT + D_ + n] = (_Float16)acc;
  }
  for (int idx = tid; idx < MT * D_; idx += 256) {   // inp0 = 0
    int m = idx >> 6, d = idx & 63;
    a_h[m * KTOT + d] = (_Float16)0.f;
  }
  if (tid < G_) bias[tid] = r_b_ih[tid] + r_b_hh[tid];
  __syncthreads();

  // ---------------- decoder scan ----------------
  for (int t = 0; t < T_; ++t) {
    gate_gemm(a_u, decW, gates, w, lane);
    __syncthreads();
    for (int idx = tid; idx < MT * H_; idx += 256) {  // LSTM cell
      int m = idx >> 7, j = idx & 127;
      const float* gm = &gates[m * G_];
      float ig = gm[j]          + bias[j];
      float fg = gm[H_ + j]     + bias[H_ + j];
      float gg = gm[2 * H_ + j] + bias[2 * H_ + j];
      float og = gm[3 * H_ + j] + bias[3 * H_ + j];
      float c = sigf(fg) * cbuf[idx] + sigf(ig) * tanhf(gg);
      float h = sigf(og) * tanhf(c);
      cbuf[idx] = c; hbuf[idx] = h;
      a_h[m * KTOT + D_ + j] = (_Float16)h;
    }
    __syncthreads();
    // out = h @ r_fc2_W^T + b (16x64, K=128) -> d_out and next input
    if (w < 4) {                       // wave-uniform branch, EXEC stays all-1s
      v8f acc = {};
#pragma unroll
      for (int ks = 0; ks < KS_OUT; ++ks) {
        v16h a = load_a_frag(a_u, lane, D_ + ks * 32);
        v16h b = load_b_frag(fc2W, w * KS_OUT + ks, lane);
        acc = __builtin_amdgcn_wmma_f32_16x16x32_f16(
            false, a, false, b, (short)0, acc, false, false);
      }
      int mhi = (lane >> 4) * 8, n = w * 16 + (lane & 15);
      float bb = r_fc2_b[n];
#pragma unroll
      for (int r = 0; r < 8; ++r) {
        float v = acc[r] + bb;
        int m = mhi + r;
        out[((size_t)(m0 + m) * T_ + t) * D_ + n] = v;
        a_h[m * KTOT + n] = (_Float16)v;   // feedback input for step t+1
      }
    }
    __syncthreads();
  }
}

// ============================================================
extern "C" void kernel_launch(void* const* d_in, const int* in_sizes, int n_in,
                              void* d_out, int out_size, void* d_ws, size_t ws_size,
                              hipStream_t stream) {
  (void)in_sizes; (void)n_in; (void)out_size; (void)ws_size;
  const float* x     = (const float*)d_in[0];
  const float* I     = (const float*)d_in[1];
  const float* eps   = (const float*)d_in[2];
  const float* gWih  = (const float*)d_in[3];
  const float* gWhh  = (const float*)d_in[4];
  const float* gbih  = (const float*)d_in[5];
  const float* gbhh  = (const float*)d_in[6];
  const float* gfcW  = (const float*)d_in[7];
  const float* gfcb  = (const float*)d_in[8];
  const float* rfc1W = (const float*)d_in[9];
  const float* rfc1b = (const float*)d_in[10];
  const float* rWih  = (const float*)d_in[11];
  const float* rWhh  = (const float*)d_in[12];
  const float* rbih  = (const float*)d_in[13];
  const float* rbhh  = (const float*)d_in[14];
  const float* rfc2W = (const float*)d_in[15];
  const float* rfc2b = (const float*)d_in[16];
  float* out = (float*)d_out;

  pack_weights<<<(WS_HALFS + 255) / 256, 256, 0, stream>>>(
      gWih, gWhh, rWih, rWhh, rfc2W, (_Float16*)d_ws);
  vrnn_main<<<NWG, 256, 0, stream>>>(
      x, I, eps, gbih, gbhh, gfcW, gfcb, rfc1W, rfc1b,
      rbih, rbhh, rfc2b, (const unsigned int*)d_ws, out);
}
